// LSTMForecaster_62955630625432
// MI455X (gfx1250) — compile-verified, matrix operand order checked by reading, not run on hardware
//
#include <hip/hip_runtime.h>
#include <cstdint>
#include <cstddef>

// ---------------------------------------------------------------------------
// Types for CDNA5 WMMA (wave32): v_wmma_f32_16x16x32_bf16
// ---------------------------------------------------------------------------
typedef __bf16 bf16_t;
typedef __attribute__((ext_vector_type(16))) __bf16 v16bf;
typedef __attribute__((ext_vector_type(8)))  __bf16 bf16x8;
typedef __attribute__((ext_vector_type(4)))  __bf16 bf16x4;
typedef __attribute__((ext_vector_type(8)))  float  v8f;

union FragBF {
    v16bf  v;
    bf16x8 h[2];
};
union PackBF2 {
    uint32_t u;
    bf16_t   h[2];
};

__device__ __forceinline__ bf16_t f2bf(float x) {
    union { float f; uint32_t u; } cv; cv.f = x;
    uint32_t u = cv.u;
    u += 0x7FFFu + ((u >> 16) & 1u);        // round-to-nearest-even
    union { unsigned short s; bf16_t b; } r;
    r.s = (unsigned short)(u >> 16);
    return r.b;
}

// ---------------------------------------------------------------------------
// bf16 WMMA GEMM:  C = A(MxK) * B(KxN) [+ Cinit] [+ bias]
// A row-major bf16, B row-major bf16, C row-major f32.
// Block tile TBM(M) x 64(N), K chunked by 32, double-buffered LDS.
// TBM*2 threads = TBM/16 waves; wave w computes rows [w*16,w*16+16) x 64 cols
// (four 16x16 wmma tiles).  Requires M%TBM==0, N%64==0, K%32==0.
//
// LDS layouts:
//   lds_a[p]: [row][k] row-major bf16, stride 32  -> A-fragment = 2x b128 load
//   lds_b[p]: [n][k/2] k-pairs packed in dwords   -> B-fragment = 2x b128 load
//
// Inner loop preloads all 4 B fragments, then issues the 4 WMMAs
// back-to-back (disjoint accumulators -> no D->A/B hazards).
// ---------------------------------------------------------------------------
#define BN 64
#define BK 32

template<int TBM>
__global__ __launch_bounds__(TBM * 2)
void gemm_bf16_wmma(const bf16_t* __restrict__ A, int lda,
                    const bf16_t* __restrict__ B, int ldb,
                    const float*  __restrict__ Cinit, long long ldc_init,
                    const float*  __restrict__ bias,
                    float*        __restrict__ C, int ldc,
                    int K)
{
    constexpr int NT    = TBM * 2;        // threads per block
    constexpr int NSLOT = 256 / NT;       // B staging slots per thread (1,2,4)

    __shared__ bf16_t   lds_a[2][TBM * BK];
    __shared__ uint32_t lds_b[2][BN * (BK / 2)];

    const int tid  = threadIdx.x;
    const int lane = tid & 31;
    const int wave = tid >> 5;
    const long long bm = (long long)blockIdx.x * TBM;
    const int       bn = blockIdx.y * BN;

    // C/D fragment mapping (16x16 f32, 8 VGPRs): n = lane&15, m = r + 8*(lane>>4)
    const int cm = (lane >> 4) << 3;
    const int cn = lane & 15;

    v8f acc[4];
#pragma unroll
    for (int s = 0; s < 4; ++s) {
        v8f a;
        const int coln = bn + s * 16 + cn;
        const float bv = bias ? bias[coln] : 0.0f;
#pragma unroll
        for (int r = 0; r < 8; ++r) {
            const long long row = bm + wave * 16 + r + cm;
            float v = bv;
            if (Cinit) v += Cinit[row * ldc_init + coln];
            a[r] = v;
        }
        acc[s] = a;
    }

    // --- staging assignments ---
    const int arow = tid >> 1;            // 0..TBM-1
    const int acol = (tid & 1) * 16;      // 0 or 16

    // A-fragment mapping (16x32 bf16): m = lane&15,
    //   elems 0..7 -> k = kbase+0..7, elems 8..15 -> k = kbase+16..23,
    //   kbase = 8*(lane>>4)
    const int am    = wave * 16 + (lane & 15);
    const int kbase = (lane >> 4) * 8;

    bf16x8 ra0, ra1;
    bf16x4 rb0[NSLOT], rb1[NSLOT];

    auto gload = [&](int kk) {
        const bf16_t* ag = A + (long long)(bm + arow) * lda + kk + acol;
        ra0 = *(const bf16x8*)(ag);
        ra1 = *(const bf16x8*)(ag + 8);
#pragma unroll
        for (int q = 0; q < NSLOT; ++q) {
            const int s   = tid + q * NT;   // 0..255
            const int kp  = s >> 4;         // k-pair index 0..15 -> k = 2*kp
            const int ncg = s & 15;         // col group of 4 -> n = 4*ncg
            const bf16_t* bg = B + (long long)(kk + kp * 2) * ldb + bn + ncg * 4;
            rb0[q] = *(const bf16x4*)(bg);
            rb1[q] = *(const bf16x4*)(bg + ldb);
        }
    };
    auto stage = [&](int p) {
        *(bf16x8*)(&lds_a[p][arow * BK + acol])     = ra0;
        *(bf16x8*)(&lds_a[p][arow * BK + acol + 8]) = ra1;
#pragma unroll
        for (int q = 0; q < NSLOT; ++q) {
            const int s   = tid + q * NT;
            const int kp  = s >> 4;
            const int ncg = s & 15;
#pragma unroll
            for (int j = 0; j < 4; ++j) {
                PackBF2 w;
                w.h[0] = rb0[q][j];
                w.h[1] = rb1[q][j];
                lds_b[p][(ncg * 4 + j) * (BK / 2) + kp] = w.u;
            }
        }
    };

    gload(0);
    stage(0);
    __syncthreads();

    int p = 0;
    for (int kk = 0; kk < K; kk += BK) {
        const bool more = (kk + BK) < K;
        if (more) gload(kk + BK);   // overlap next-chunk global loads with WMMA

        // Preload A fragment and ALL four B fragments, then issue the four
        // WMMAs back-to-back (single DS wait instead of one per subtile).
        FragBF fa, fb[4];
        fa.h[0] = *(const bf16x8*)(&lds_a[p][am * BK + kbase]);
        fa.h[1] = *(const bf16x8*)(&lds_a[p][am * BK + kbase + 16]);
#pragma unroll
        for (int s = 0; s < 4; ++s) {
            const int bnn = s * 16 + (lane & 15);
            fb[s].h[0] = *(const bf16x8*)(&lds_b[p][bnn * (BK / 2) + (kbase >> 1)]);
            fb[s].h[1] = *(const bf16x8*)(&lds_b[p][bnn * (BK / 2) + 8 + (kbase >> 1)]);
        }
#pragma unroll
        for (int s = 0; s < 4; ++s) {
            acc[s] = __builtin_amdgcn_wmma_f32_16x16x32_bf16(
                         false, fa.v, false, fb[s].v,
                         (short)0, acc[s], false, false);
        }

        if (more) {
            stage(p ^ 1);
            __syncthreads();
            p ^= 1;
        }
    }

#pragma unroll
    for (int s = 0; s < 4; ++s) {
        const int coln = bn + s * 16 + cn;
#pragma unroll
        for (int r = 0; r < 8; ++r) {
            const long long row = bm + wave * 16 + r + cm;
            C[row * (long long)ldc + coln] = acc[s][r];
        }
    }
}

// ---------------------------------------------------------------------------
// LSTM pointwise cell: z:(B,4H) -> update c,h ; emit bf16 h for next GEMM.
// Gate order (flax): i, f, g, o in columns [0,H),[H,2H),[2H,3H),[3H,4H).
// ---------------------------------------------------------------------------
__global__ __launch_bounds__(256)
void lstm_cell(const float* __restrict__ z,
               float* __restrict__ c, float* __restrict__ h,
               bf16_t* __restrict__ hbf,
               bf16_t* __restrict__ out_all, long long out_stride_b,
               int B, int H)
{
    const int idx = blockIdx.x * blockDim.x + threadIdx.x;
    if (idx >= B * H) return;
    const int b = idx / H;
    const int j = idx - b * H;
    const float* zb = z + (long long)b * 4 * H;

    float vi = zb[j];
    float vf = zb[H + j];
    float vg = zb[2 * H + j];
    float vo = zb[3 * H + j];

    vi = 1.0f / (1.0f + __expf(-vi));
    vf = 1.0f / (1.0f + __expf(-vf));
    vg = tanhf(vg);
    vo = 1.0f / (1.0f + __expf(-vo));

    const float cv = vf * c[idx] + vi * vg;
    const float hv = vo * tanhf(cv);
    c[idx] = cv;
    h[idx] = hv;
    const bf16_t hb = f2bf(hv);
    hbf[idx] = hb;
    if (out_all) out_all[(long long)b * out_stride_b + j] = hb;
}

// ---------------------------------------------------------------------------
// f32 -> bf16 conversion
// ---------------------------------------------------------------------------
__global__ __launch_bounds__(256)
void cvt_f32_bf16(const float* __restrict__ in, bf16_t* __restrict__ out,
                  long long n)
{
    const long long i = (long long)blockIdx.x * blockDim.x + threadIdx.x;
    if (i < n) out[i] = f2bf(in[i]);
}

// ---------------------------------------------------------------------------
// zero dwords
// ---------------------------------------------------------------------------
__global__ __launch_bounds__(256)
void zero_dwords(uint32_t* __restrict__ p, long long n)
{
    const long long i = (long long)blockIdx.x * blockDim.x + threadIdx.x;
    if (i < n) p[i] = 0u;
}

// ---------------------------------------------------------------------------
// Final FC: out(B,O) = h(B,H) @ Wfc(H,O) + bfc   (tiny: 256x64x512)
// ---------------------------------------------------------------------------
__global__ __launch_bounds__(256)
void fc_kernel(const float* __restrict__ h, const float* __restrict__ W,
               const float* __restrict__ bias, float* __restrict__ out,
               int B, int H, int O)
{
    const int idx = blockIdx.x * blockDim.x + threadIdx.x;
    if (idx >= B * O) return;
    const int b = idx / O;
    const int o = idx - b * O;
    float s = bias[o];
    const float* hb = h + (long long)b * H;
    for (int k = 0; k < H; ++k)
        s += hb[k] * W[(long long)k * O + o];
    out[idx] = s;
}

// ---------------------------------------------------------------------------
// Host-side orchestration
// ---------------------------------------------------------------------------
extern "C" void kernel_launch(void* const* d_in, const int* in_sizes, int n_in,
                              void* d_out, int out_size, void* d_ws, size_t ws_size,
                              hipStream_t stream)
{
    (void)in_sizes; (void)n_in; (void)out_size; (void)ws_size;

    constexpr int Bb = 256, Ss = 512, Ff = 128, Hh = 512, Oo = 64;
    constexpr int Gg = 4 * Hh; // 2048

    const float* x   = (const float*)d_in[0];
    const float* Wi0 = (const float*)d_in[1];
    const float* Wh0 = (const float*)d_in[2];
    const float* b0  = (const float*)d_in[3];
    const float* Wi1 = (const float*)d_in[4];
    const float* Wh1 = (const float*)d_in[5];
    const float* b1  = (const float*)d_in[6];
    const float* Wfc = (const float*)d_in[7];
    const float* bfc = (const float*)d_in[8];
    float* out = (float*)d_out;

    // ---- workspace carve (all chunks 256B-aligned; sizes already 256-mult) --
    char* ws = (char*)d_ws;
    size_t off = 0;
    auto carve = [&](size_t bytes) -> char* {
        char* p = ws + off;
        off += (bytes + 255) & ~(size_t)255;
        return p;
    };
    float*  xW   = (float*) carve((size_t)Bb * Ss * Gg * sizeof(float));   // 1 GiB
    bf16_t* h1   = (bf16_t*)carve((size_t)Bb * Ss * Hh * sizeof(bf16_t)); // 128 MiB
    bf16_t* xbf  = (bf16_t*)carve((size_t)Bb * Ss * Ff * sizeof(bf16_t)); // 32 MiB
    bf16_t* wi0b = (bf16_t*)carve((size_t)Ff * Gg * sizeof(bf16_t));
    bf16_t* wh0b = (bf16_t*)carve((size_t)Hh * Gg * sizeof(bf16_t));
    bf16_t* wi1b = (bf16_t*)carve((size_t)Hh * Gg * sizeof(bf16_t));
    bf16_t* wh1b = (bf16_t*)carve((size_t)Hh * Gg * sizeof(bf16_t));
    float*  zbuf = (float*) carve((size_t)Bb * Gg * sizeof(float));
    // c, h, hbf kept contiguous so one zero_dwords covers all state
    float*  cst  = (float*) carve((size_t)Bb * Hh * sizeof(float));
    float*  hst  = (float*) carve((size_t)Bb * Hh * sizeof(float));
    bf16_t* hbf  = (bf16_t*)carve((size_t)Bb * Hh * sizeof(bf16_t));

    // ---- weight / input conversions to bf16 ----
    {
        long long n = (long long)Bb * Ss * Ff;
        cvt_f32_bf16<<<dim3((unsigned)((n + 255) / 256)), 256, 0, stream>>>(x, xbf, n);
        n = (long long)Ff * Gg;
        cvt_f32_bf16<<<dim3((unsigned)((n + 255) / 256)), 256, 0, stream>>>(Wi0, wi0b, n);
        n = (long long)Hh * Gg;
        cvt_f32_bf16<<<dim3((unsigned)((n + 255) / 256)), 256, 0, stream>>>(Wh0, wh0b, n);
        cvt_f32_bf16<<<dim3((unsigned)((n + 255) / 256)), 256, 0, stream>>>(Wi1, wi1b, n);
        cvt_f32_bf16<<<dim3((unsigned)((n + 255) / 256)), 256, 0, stream>>>(Wh1, wh1b, n);
    }

    // Big GEMMs: 128-row tiles, 256 threads.  Step GEMMs: 64-row tiles,
    // 128 threads -> 128 blocks (better WGP coverage in the sequential part).
    const dim3 gridBig((Bb * Ss) / 128, Gg / BN);   // 1024 x 32
    const dim3 gridStep(Bb / 64, Gg / BN);          // 4 x 32
    const long long stateDwords =
        ((long long)Bb * Hh * 4 * 2 + (long long)Bb * Hh * 2) / 4;
    const int cellBlocks = (Bb * Hh + 255) / 256;

    // ===================== Layer 0 =====================
    // xW = x @ Wi0 + b0   (big GEMM, whole sequence at once)
    gemm_bf16_wmma<128><<<gridBig, 256, 0, stream>>>(
        xbf, Ff, wi0b, Gg, nullptr, 0, b0, xW, Gg, Ff);

    zero_dwords<<<dim3((unsigned)((stateDwords + 255) / 256)), 256, 0, stream>>>(
        (uint32_t*)cst, stateDwords);

    for (int t = 0; t < Ss; ++t) {
        // z = xW[:,t,:] + h @ Wh0   (bias already folded into xW)
        gemm_bf16_wmma<64><<<gridStep, 128, 0, stream>>>(
            hbf, Hh, wh0b, Gg,
            xW + (size_t)t * Gg, (long long)Ss * Gg,
            nullptr, zbuf, Gg, Hh);
        lstm_cell<<<cellBlocks, 256, 0, stream>>>(
            zbuf, cst, hst, hbf,
            h1 + (size_t)t * Hh, (long long)Ss * Hh, Bb, Hh);
    }

    // ===================== Layer 1 =====================
    // xW = h1 @ Wi1 + b1  (big GEMM over whole layer-0 output)
    gemm_bf16_wmma<128><<<gridBig, 256, 0, stream>>>(
        h1, Hh, wi1b, Gg, nullptr, 0, b1, xW, Gg, Hh);

    zero_dwords<<<dim3((unsigned)((stateDwords + 255) / 256)), 256, 0, stream>>>(
        (uint32_t*)cst, stateDwords);

    for (int t = 0; t < Ss; ++t) {
        gemm_bf16_wmma<64><<<gridStep, 128, 0, stream>>>(
            hbf, Hh, wh1b, Gg,
            xW + (size_t)t * Gg, (long long)Ss * Gg,
            nullptr, zbuf, Gg, Hh);
        lstm_cell<<<cellBlocks, 256, 0, stream>>>(
            zbuf, cst, hst, hbf,
            nullptr, 0, Bb, Hh);
    }

    // ===================== FC head =====================
    fc_kernel<<<dim3((Bb * Oo + 255) / 256), 256, 0, stream>>>(
        hst, Wfc, bfc, out, Bb, Hh, Oo);
}
